// VQVAE1D_2130303779190
// MI455X (gfx1250) — matrix operand order repaired and checked by reading, not compiled, not run on hardware
//
#include <hip/hip_runtime.h>
#include <hip/hip_bf16.h>

typedef __attribute__((ext_vector_type(16))) _Float16 v16h;
typedef __attribute__((ext_vector_type(8)))  _Float16 v8h;
typedef __attribute__((ext_vector_type(8)))  float    v8f;

#define BATCH 32
#define TLEN  2048
#define KLAT  512
#define HID   256
#define CDIM  128
#define NCODE 1024

// ---------------------------------------------------------------------------
// Weight packing: w (COUT, CIN, K) fp32 -> (COUT, K*CIN) f16, c = kk*CIN + ic
// ---------------------------------------------------------------------------
__global__ void pack_oik(const float* __restrict__ w, _Float16* __restrict__ out,
                         int COUT, int CIN, int K) {
    int idx = blockIdx.x * 256 + threadIdx.x;
    int CK = CIN * K;
    if (idx >= COUT * CK) return;
    int oc = idx / CK, c = idx - oc * CK;
    int kk = c / CIN, ic = c - kk * CIN;
    out[idx] = (_Float16)w[((size_t)oc * CIN + ic) * K + kk];
}

// up_w (IN=256, OUT=256, 8) -> 4 phase convs: out (4, 256, 512), c = kk*256+ic
// phase p: k0=(p+2)&3; kk=0 -> tap k0+4 (src t-1+d0), kk=1 -> tap k0 (src t+d0)
__global__ void pack_up(const float* __restrict__ w, _Float16* __restrict__ out) {
    int idx = blockIdx.x * 256 + threadIdx.x;
    if (idx >= 4 * 256 * 512) return;
    int p = idx / (256 * 512);
    int r = idx - p * 256 * 512;
    int oc = r / 512, c = r - oc * 512;
    int kk = c / 256, ic = c - kk * 256;
    int k0 = (p + 2) & 3;
    int k = (kk == 0) ? (k0 + 4) : k0;
    out[idx] = (_Float16)w[((size_t)ic * 256 + oc) * 8 + k];
}

// dec_out_w (36,256,3) -> (48, 768) zero-padded
__global__ void pack_decout(const float* __restrict__ w, _Float16* __restrict__ out) {
    int idx = blockIdx.x * 256 + threadIdx.x;
    if (idx >= 48 * 768) return;
    int oc = idx / 768, c = idx - oc * 768;
    int kk = c / 256, ic = c - kk * 256;
    out[idx] = (oc < 36) ? (_Float16)w[((size_t)oc * 256 + ic) * 3 + kk] : (_Float16)0.f;
}

// ---------------------------------------------------------------------------
// GroupNorm stats: X (B,256,T), 8 groups of 32 channels -> mu, rstd (B,8)
// ---------------------------------------------------------------------------
__global__ __launch_bounds__(256) void gn_stats(const float* __restrict__ X,
                                                float* __restrict__ mu,
                                                float* __restrict__ rstd, int T) {
    int g = blockIdx.x, b = blockIdx.y, tid = threadIdx.x;
    const float* base = X + ((size_t)b * 256 + g * 32) * T;
    float s = 0.f, s2 = 0.f;
    for (int c = 0; c < 32; ++c) {
        const float* row = base + (size_t)c * T;
        for (int t = tid; t < T; t += 256) { float v = row[t]; s += v; s2 += v * v; }
    }
    __shared__ float sh[256], sh2[256];
    sh[tid] = s; sh2[tid] = s2; __syncthreads();
    for (int o = 128; o > 0; o >>= 1) {
        if (tid < o) { sh[tid] += sh[tid + o]; sh2[tid] += sh2[tid + o]; }
        __syncthreads();
    }
    if (tid == 0) {
        float n = 32.f * (float)T;
        float m = sh[0] / n;
        float var = sh2[0] / n - m * m;
        mu[b * 8 + g] = m;
        rstd[b * 8 + g] = rsqrtf(var + 1e-5f);
    }
}

// ---------------------------------------------------------------------------
// Input conv: x (B,T,36) -> (B,256,T), k=3 pad=1 (tiny Cin, scalar)
// ---------------------------------------------------------------------------
__global__ __launch_bounds__(256) void conv_in_kernel(const float* __restrict__ x,
                                                      const float* __restrict__ w,
                                                      const float* __restrict__ bias,
                                                      float* __restrict__ out) {
    size_t idx = (size_t)blockIdx.x * 256 + threadIdx.x;  // B*256*T
    int t = (int)(idx % TLEN);
    int rest = (int)(idx / TLEN);
    int oc = rest & 255;
    int b = rest >> 8;
    float acc = bias[oc];
    for (int k = 0; k < 3; ++k) {
        int ts = t + k - 1;
        if (ts < 0 || ts >= TLEN) continue;
        const float* xr = x + ((size_t)b * TLEN + ts) * 36;
        const float* wr = w + (size_t)oc * 36 * 3 + k;
        #pragma unroll 4
        for (int ic = 0; ic < 36; ++ic) acc += wr[ic * 3] * xr[ic];
    }
    out[idx] = acc;
}

// ---------------------------------------------------------------------------
// Generic WMMA conv:  Y[b][oc][out_base + OSTR*t] =
//   bias[oc] (+ res) + sum_{kk,ic} Wp[oc][kk*CIN+ic] * pre(X[b][ic][t*SIN+kk-PAD])
// pre = GN+SiLU when PRE_GN.  XHAT_OUT stores to (B,T,36) layout.
// Block: 256 threads = 8 waves; each block does one (b, NT*16-column) tile.
// NT column sub-tiles share one A fragment per K-step (A reuse, paired WMMAs).
// All shape parameters are compile-time so the epilogue is branch-free with
// constant-strength address math (except the 36-of-48 masked output conv).
// ---------------------------------------------------------------------------
template <int CIN, int KT, int SIN, int PRE_GN, int ADD_RES, int XHAT_OUT,
          int COUT, int CVALID, int TTOT, int OSTR, int NT>
__global__ __launch_bounds__(256)
void wmma_conv(const float* __restrict__ X, const _Float16* __restrict__ Wp,
               const float* __restrict__ bias, const float* __restrict__ res,
               const float* __restrict__ gam, const float* __restrict__ bet,
               const float* __restrict__ mu, const float* __restrict__ rstd,
               float* __restrict__ Y, int Ti, int PAD, int out_base) {
    constexpr int CK = CIN * KT;          // contraction length (multiple of 32)
    constexpr int NCOL = 16 * NT;         // output columns per block
    constexpr int NSH = (NT == 2) ? 5 : 4;
    __shared__ _Float16 Bls[NCOL][CK];
    const int tid = threadIdx.x;
    const int b = blockIdx.y;
    const int t0 = blockIdx.x * NCOL;
    const float* Xb = X + (size_t)b * CIN * Ti;

    // ---- stage B tile (im2col) into LDS, fused GN+SiLU ----
    for (int idx = tid; idx < NCOL * CK; idx += 256) {
        int n = idx & (NCOL - 1);      // column (time) -> coalesced over threads
        int c = idx >> NSH;            // contraction index
        int kk = c / CIN;
        int ic = c - kk * CIN;
        int ts = (t0 + n) * SIN + kk - PAD;
        float v = 0.f;
        if (ts >= 0 && ts < Ti) v = Xb[(size_t)ic * Ti + ts];
        if (PRE_GN) {  // CIN==256, 8 groups of 32
            int g = ic >> 5;
            float m = mu[b * 8 + g], rs = rstd[b * 8 + g];
            v = (v - m) * rs * gam[ic] + bet[ic];
            v = v / (1.f + __expf(-v));          // SiLU
        }
        Bls[n][c] = (_Float16)v;
    }
    __syncthreads();

    const int lane = tid & 31;
    const int wv = tid >> 5;
    const int hi = lane >> 4;       // half-wave select
    const int lm = lane & 15;
    constexpr int NTILES = COUT >> 4;

    for (int ot = wv; ot < NTILES; ot += 8) {
        int oc0 = ot * 16;
        v8f acc[NT] = {};
        // A: lane (hi,lm) holds row M=lm; element e -> K=e+hi*16
        const _Float16* wrow = Wp + (size_t)(oc0 + lm) * CK + hi * 16;
        // B: lane (hi,lm) holds col N=lm; element e -> K=e+hi*16
        const _Float16* brow[NT];
        #pragma unroll
        for (int j = 0; j < NT; ++j) brow[j] = &Bls[j * 16 + lm][hi * 16];

        #pragma unroll 4
        for (int s = 0; s < CK / 32; ++s) {
            v8h alo = *(const v8h*)(wrow + s * 32);
            v8h ahi = *(const v8h*)(wrow + s * 32 + 8);
            v16h a = __builtin_shufflevector(alo, ahi, 0,1,2,3,4,5,6,7,8,9,10,11,12,13,14,15);
            #pragma unroll
            for (int j = 0; j < NT; ++j) {
                v8h blo = *(const v8h*)(brow[j] + s * 32);
                v8h bhi = *(const v8h*)(brow[j] + s * 32 + 8);
                v16h bm = __builtin_shufflevector(blo, bhi, 0,1,2,3,4,5,6,7,8,9,10,11,12,13,14,15);
                acc[j] = __builtin_amdgcn_wmma_f32_16x16x32_f16(false, a, false, bm,
                                                                (short)0, acc[j], false, false);
            }
        }
        // D: lane (hi,lm): VGPR r -> (M = r + hi*8, N = lm)
        #pragma unroll
        for (int j = 0; j < NT; ++j) {
            int tout = out_base + OSTR * (t0 + j * 16 + lm);
            #pragma unroll
            for (int r2 = 0; r2 < 8; ++r2) {
                int oc = oc0 + r2 + hi * 8;
                if (COUT == CVALID || oc < CVALID) {
                    float v = acc[j][r2] + bias[oc];
                    if (XHAT_OUT) {
                        Y[((size_t)b * TTOT + tout) * 36 + oc] = v;
                    } else {
                        size_t o = ((size_t)b * COUT + oc) * TTOT + tout;
                        if (ADD_RES) v += res[o];
                        Y[o] = v;
                    }
                }
            }
        }
    }
}

// ---------------------------------------------------------------------------
// VQ
// ---------------------------------------------------------------------------
__global__ __launch_bounds__(128) void cnorm_kernel(const float* __restrict__ cb,
                                                    float* __restrict__ cn) {
    int code = blockIdx.x, tid = threadIdx.x;
    __shared__ float sh[128];
    float v = cb[(size_t)code * 128 + tid];
    sh[tid] = v * v; __syncthreads();
    for (int o = 64; o > 0; o >>= 1) { if (tid < o) sh[tid] += sh[tid + o]; __syncthreads(); }
    if (tid == 0) cn[code] = sh[0];
}

__global__ __launch_bounds__(128)
void vq_kernel(const float* __restrict__ ze, const float* __restrict__ cb,
               const float* __restrict__ cnorm, float* __restrict__ zq,
               float* __restrict__ codes_f, float* __restrict__ vq_part,
               int* __restrict__ counts) {
    const int t = blockIdx.x, b = blockIdx.y, tid = threadIdx.x;
    __shared__ float zsh[128];
    __shared__ float dsh[128];
    __shared__ int ish[128];
    __shared__ int bestsh;
    zsh[tid] = ze[((size_t)b * 128 + tid) * KLAT + t];
    __syncthreads();
    float best = 3.4e38f; int bi = 0;
    for (int j = 0; j < 8; ++j) {
        int code = tid + 128 * j;
        const float* cr = cb + (size_t)code * 128;
        float dot = 0.f;
        #pragma unroll 8
        for (int c = 0; c < 128; ++c) dot += zsh[c] * cr[c];
        float d = cnorm[code] - 2.f * dot;                 // |z|^2 constant per row
        if (d < best || (d == best && code < bi)) { best = d; bi = code; }
    }
    dsh[tid] = best; ish[tid] = bi; __syncthreads();
    for (int o = 64; o > 0; o >>= 1) {
        if (tid < o) {
            float d2 = dsh[tid + o]; int i2 = ish[tid + o];
            if (d2 < dsh[tid] || (d2 == dsh[tid] && i2 < ish[tid])) { dsh[tid] = d2; ish[tid] = i2; }
        }
        __syncthreads();
    }
    if (tid == 0) {
        bestsh = ish[0];
        codes_f[(size_t)b * KLAT + t] = (float)ish[0];
        atomicAdd(&counts[ish[0]], 1);
    }
    __syncthreads();
    int code = bestsh;
    float qv = cb[(size_t)code * 128 + tid];
    zq[((size_t)b * 128 + tid) * KLAT + t] = qv;
    float df = qv - zsh[tid];
    dsh[tid] = df * df; __syncthreads();
    for (int o = 64; o > 0; o >>= 1) { if (tid < o) dsh[tid] += dsh[tid + o]; __syncthreads(); }
    if (tid == 0) vq_part[(size_t)b * KLAT + t] = dsh[0];
}

__global__ __launch_bounds__(256) void vq_finalize(const float* __restrict__ vq_part,
                                                   const int* __restrict__ counts,
                                                   float* __restrict__ scal) {
    __shared__ float sh[256];
    int tid = threadIdx.x;
    float s = 0.f;
    for (int i = tid; i < BATCH * KLAT; i += 256) s += vq_part[i];
    sh[tid] = s; __syncthreads();
    for (int o = 128; o > 0; o >>= 1) { if (tid < o) sh[tid] += sh[tid + o]; __syncthreads(); }
    if (tid == 0) scal[0] = sh[0] / (float)(BATCH * 128 * KLAT) * 1.25f;  // (1+beta)*mse
    __syncthreads();
    float h = 0.f;
    for (int i = tid; i < NCODE; i += 256) {
        float avg = (float)counts[i] / (float)(BATCH * KLAT);
        h += avg * logf(avg + 1e-10f);
    }
    sh[tid] = h; __syncthreads();
    for (int o = 128; o > 0; o >>= 1) { if (tid < o) sh[tid] += sh[tid + o]; __syncthreads(); }
    if (tid == 0) scal[1] = expf(-sh[0]);
}

// ---------------------------------------------------------------------------
// Reconstruction losses
// ---------------------------------------------------------------------------
__global__ __launch_bounds__(256) void loss_rows(const float* __restrict__ x,
                                                 const float* __restrict__ xh,
                                                 const float* __restrict__ mask,
                                                 float* __restrict__ part) {
    int i = blockIdx.x * 256 + threadIdx.x;   // one (b,t) row; 65536 total
    const float* xr = x + (size_t)i * 36;
    const float* hr = xh + (size_t)i * 36;
    float pos = 0.f;
    #pragma unroll
    for (int j = 0; j < 21; ++j) {
        float d = hr[j] - xr[j]; float ad = fabsf(d);
        pos += (ad < 1.f) ? 0.5f * d * d : ad - 0.5f;
    }
    #pragma unroll
    for (int j = 27; j < 30; ++j) {
        float d = hr[j] - xr[j]; float ad = fabsf(d);
        pos += (ad < 1.f) ? 0.5f * d * d : ad - 0.5f;
    }
    float mL = mask[(size_t)i * 2], mR = mask[(size_t)i * 2 + 1];
    const int st[4] = {21, 24, 30, 33};
    const float mv[4] = {mL, mL, mR, mR};
    float dn = 0.f, dd = 0.f;
    #pragma unroll
    for (int v = 0; v < 4; ++v) {
        int s0 = st[v];
        float ax = xr[s0], ay = xr[s0 + 1], az = xr[s0 + 2];
        float bx = hr[s0], by = hr[s0 + 1], bz = hr[s0 + 2];
        float na = sqrtf(ax * ax + ay * ay + az * az) + 1e-8f;
        float nb = sqrtf(bx * bx + by * by + bz * bz) + 1e-8f;
        float cs = (ax * bx + ay * by + az * bz) / (na * nb);
        dn += (1.f - cs) * mv[v];
        dd += mv[v];
    }
    __shared__ float s0[256], s1[256], s2[256];
    int tid = threadIdx.x;
    s0[tid] = pos; s1[tid] = dn; s2[tid] = dd; __syncthreads();
    for (int o = 128; o > 0; o >>= 1) {
        if (tid < o) { s0[tid] += s0[tid + o]; s1[tid] += s1[tid + o]; s2[tid] += s2[tid + o]; }
        __syncthreads();
    }
    if (tid == 0) {
        part[blockIdx.x * 3 + 0] = s0[0];
        part[blockIdx.x * 3 + 1] = s1[0];
        part[blockIdx.x * 3 + 2] = s2[0];
    }
}

__global__ __launch_bounds__(256) void loss_final(const float* __restrict__ part,
                                                  float* __restrict__ scal) {
    __shared__ float s0[256], s1[256], s2[256];
    int tid = threadIdx.x;
    s0[tid] = part[tid * 3 + 0];
    s1[tid] = part[tid * 3 + 1];
    s2[tid] = part[tid * 3 + 2];
    __syncthreads();
    for (int o = 128; o > 0; o >>= 1) {
        if (tid < o) { s0[tid] += s0[tid + o]; s1[tid] += s1[tid + o]; s2[tid] += s2[tid + o]; }
        __syncthreads();
    }
    if (tid == 0) {
        scal[2] = s0[0] / (float)(BATCH * TLEN);
        scal[3] = s1[0] / fmaxf(s2[0], 1.f);
    }
}

// ---------------------------------------------------------------------------
// Host driver
// ---------------------------------------------------------------------------
extern "C" void kernel_launch(void* const* d_in, const int* in_sizes, int n_in,
                              void* d_out, int out_size, void* d_ws, size_t ws_size,
                              hipStream_t stream) {
    const float* x          = (const float*)d_in[0];
    const float* hand_mask  = (const float*)d_in[1];
    const float* enc_in_w   = (const float*)d_in[2];
    const float* enc_in_b   = (const float*)d_in[3];
    const float* enc_gn1_g  = (const float*)d_in[4];
    const float* enc_gn1_b  = (const float*)d_in[5];
    const float* enc_c1_w   = (const float*)d_in[6];
    const float* enc_c1_b   = (const float*)d_in[7];
    const float* enc_gn2_g  = (const float*)d_in[8];
    const float* enc_gn2_b  = (const float*)d_in[9];
    const float* enc_c2_w   = (const float*)d_in[10];
    const float* enc_c2_b   = (const float*)d_in[11];
    const float* down_w     = (const float*)d_in[12];
    const float* down_b     = (const float*)d_in[13];
    const float* enc_out_w  = (const float*)d_in[14];
    const float* enc_out_b  = (const float*)d_in[15];
    const float* codebook   = (const float*)d_in[16];
    const float* dec_in_w   = (const float*)d_in[17];
    const float* dec_in_b   = (const float*)d_in[18];
    const float* up_w       = (const float*)d_in[19];
    const float* up_b       = (const float*)d_in[20];
    const float* dec_gn1_g  = (const float*)d_in[21];
    const float* dec_gn1_b  = (const float*)d_in[22];
    const float* dec_c1_w   = (const float*)d_in[23];
    const float* dec_c1_b   = (const float*)d_in[24];
    const float* dec_gn2_g  = (const float*)d_in[25];
    const float* dec_gn2_b  = (const float*)d_in[26];
    const float* dec_c2_w   = (const float*)d_in[27];
    const float* dec_c2_b   = (const float*)d_in[28];
    const float* dec_out_w  = (const float*)d_in[29];
    const float* dec_out_b  = (const float*)d_in[30];

    float* out = (float*)d_out;
    const size_t XHAT_OFF = 0;
    const size_t CODES_OFF = (size_t)BATCH * TLEN * 36;          // 2359296
    const size_t SCAL_OFF = CODES_OFF + (size_t)BATCH * KLAT;    // +16384

    char* ws = (char*)d_ws;
    size_t off = 0;
    auto take = [&](size_t bytes) -> char* {
        char* p = ws + off;
        off += (bytes + 255) & ~(size_t)255;
        return p;
    };
    const size_t BIG = (size_t)BATCH * HID * TLEN * 4;   // 64 MB
    float* buf0  = (float*)take(BIG);
    float* buf1  = (float*)take(BIG);
    float* buf2  = (float*)take(BIG);
    float* ze    = (float*)take((size_t)BATCH * CDIM * KLAT * 4);
    float* zq    = (float*)take((size_t)BATCH * CDIM * KLAT * 4);
    float* muv   = (float*)take(BATCH * 8 * 4);
    float* rstdv = (float*)take(BATCH * 8 * 4);
    float* cnorm = (float*)take(NCODE * 4);
    int*   counts= (int*)  take(NCODE * 4);
    float* vqp   = (float*)take((size_t)BATCH * KLAT * 4);
    float* lpart = (float*)take(256 * 3 * 4);
    _Float16* wenc1   = (_Float16*)take((size_t)4 * 256 * 768 * 2);
    _Float16* wenc2   = (_Float16*)take((size_t)4 * 256 * 768 * 2);
    _Float16* wdec1   = (_Float16*)take((size_t)4 * 256 * 768 * 2);
    _Float16* wdec2   = (_Float16*)take((size_t)4 * 256 * 768 * 2);
    _Float16* wdown   = (_Float16*)take((size_t)256 * 2048 * 2);
    _Float16* wencout = (_Float16*)take((size_t)128 * 256 * 2);
    _Float16* wdecin  = (_Float16*)take((size_t)256 * 128 * 2);
    _Float16* wup     = (_Float16*)take((size_t)4 * 256 * 512 * 2);
    _Float16* wdecout = (_Float16*)take((size_t)48 * 768 * 2);
    (void)ws_size; (void)in_sizes; (void)n_in; (void)out_size;

    // ---- pack weights to f16 ----
    pack_oik<<<(1024 * 768 + 255) / 256, 256, 0, stream>>>(enc_c1_w, wenc1, 1024, 256, 3);
    pack_oik<<<(1024 * 768 + 255) / 256, 256, 0, stream>>>(enc_c2_w, wenc2, 1024, 256, 3);
    pack_oik<<<(1024 * 768 + 255) / 256, 256, 0, stream>>>(dec_c1_w, wdec1, 1024, 256, 3);
    pack_oik<<<(1024 * 768 + 255) / 256, 256, 0, stream>>>(dec_c2_w, wdec2, 1024, 256, 3);
    pack_oik<<<(256 * 2048 + 255) / 256, 256, 0, stream>>>(down_w, wdown, 256, 256, 8);
    pack_oik<<<(128 * 256 + 255) / 256, 256, 0, stream>>>(enc_out_w, wencout, 128, 256, 1);
    pack_oik<<<(256 * 128 + 255) / 256, 256, 0, stream>>>(dec_in_w, wdecin, 256, 128, 1);
    pack_up<<<(4 * 256 * 512 + 255) / 256, 256, 0, stream>>>(up_w, wup);
    pack_decout<<<(48 * 768 + 255) / 256, 256, 0, stream>>>(dec_out_w, wdecout);
    cnorm_kernel<<<NCODE, 128, 0, stream>>>(codebook, cnorm);
    hipMemsetAsync(counts, 0, NCODE * 4, stream);

    // ---- encoder ----
    conv_in_kernel<<<(BATCH * HID * TLEN) / 256, 256, 0, stream>>>(x, enc_in_w, enc_in_b, buf0);

    float* P = buf0; float* Q = buf1; float* R = buf2;
    for (int i = 0; i < 4; ++i) {
        gn_stats<<<dim3(8, BATCH), 256, 0, stream>>>(P, muv, rstdv, TLEN);
        wmma_conv<256, 3, 1, 1, 0, 0, 256, 256, TLEN, 1, 2>
            <<<dim3(TLEN / 32, BATCH), 256, 0, stream>>>(
            P, wenc1 + (size_t)i * 256 * 768, enc_c1_b + i * 256, nullptr,
            enc_gn1_g + i * 256, enc_gn1_b + i * 256, muv, rstdv, Q, TLEN, 1, 0);
        gn_stats<<<dim3(8, BATCH), 256, 0, stream>>>(Q, muv, rstdv, TLEN);
        wmma_conv<256, 3, 1, 1, 1, 0, 256, 256, TLEN, 1, 2>
            <<<dim3(TLEN / 32, BATCH), 256, 0, stream>>>(
            Q, wenc2 + (size_t)i * 256 * 768, enc_c2_b + i * 256, P,
            enc_gn2_g + i * 256, enc_gn2_b + i * 256, muv, rstdv, R, TLEN, 1, 0);
        float* t = P; P = R; R = Q; Q = t;   // (P,Q,R) <- (R,P,Q)
    }
    // down conv: k=8, stride=4, pad=2 :  P (B,256,2048) -> Q (B,256,512)
    wmma_conv<256, 8, 4, 0, 0, 0, 256, 256, KLAT, 1, 1>
        <<<dim3(KLAT / 16, BATCH), 256, 0, stream>>>(
        P, wdown, down_b, nullptr, nullptr, nullptr, nullptr, nullptr, Q, TLEN, 2, 0);
    // enc_out 1x1: Q -> z_e (B,128,512)
    wmma_conv<256, 1, 1, 0, 0, 0, 128, 128, KLAT, 1, 2>
        <<<dim3(KLAT / 32, BATCH), 256, 0, stream>>>(
        Q, wencout, enc_out_b, nullptr, nullptr, nullptr, nullptr, nullptr, ze, KLAT, 0, 0);

    // ---- VQ ----
    vq_kernel<<<dim3(KLAT, BATCH), 128, 0, stream>>>(ze, codebook, cnorm, zq,
                                                     out + CODES_OFF, vqp, counts);
    vq_finalize<<<1, 256, 0, stream>>>(vqp, counts, out + SCAL_OFF);

    // ---- decoder ----
    // dec_in 1x1: z_q (B,128,512) -> R (B,256,512)
    wmma_conv<128, 1, 1, 0, 0, 0, 256, 256, KLAT, 1, 2>
        <<<dim3(KLAT / 32, BATCH), 256, 0, stream>>>(
        zq, wdecin, dec_in_b, nullptr, nullptr, nullptr, nullptr, nullptr, R, KLAT, 0, 0);
    // transpose-up (k=8, stride=4, pad=2) as 4 phase-wise 2-tap convs: R -> P (B,256,2048)
    for (int p = 0; p < 4; ++p) {
        int d0 = (p + 2) >> 2;   // 0 for p<2, 1 for p>=2
        wmma_conv<256, 2, 1, 0, 0, 0, 256, 256, TLEN, 4, 2>
            <<<dim3(KLAT / 32, BATCH), 256, 0, stream>>>(
            R, wup + (size_t)p * 256 * 512, up_b, nullptr, nullptr, nullptr, nullptr, nullptr,
            P, KLAT, 1 - d0, p);
    }
    for (int i = 0; i < 4; ++i) {
        gn_stats<<<dim3(8, BATCH), 256, 0, stream>>>(P, muv, rstdv, TLEN);
        wmma_conv<256, 3, 1, 1, 0, 0, 256, 256, TLEN, 1, 2>
            <<<dim3(TLEN / 32, BATCH), 256, 0, stream>>>(
            P, wdec1 + (size_t)i * 256 * 768, dec_c1_b + i * 256, nullptr,
            dec_gn1_g + i * 256, dec_gn1_b + i * 256, muv, rstdv, Q, TLEN, 1, 0);
        gn_stats<<<dim3(8, BATCH), 256, 0, stream>>>(Q, muv, rstdv, TLEN);
        wmma_conv<256, 3, 1, 1, 1, 0, 256, 256, TLEN, 1, 2>
            <<<dim3(TLEN / 32, BATCH), 256, 0, stream>>>(
            Q, wdec2 + (size_t)i * 256 * 768, dec_c2_b + i * 256, P,
            dec_gn2_g + i * 256, dec_gn2_b + i * 256, muv, rstdv, R, TLEN, 1, 0);
        float* t = P; P = R; R = Q; Q = t;
    }
    // dec_out conv3 (256 -> 36, padded to 48) straight into d_out (B,T,36)
    wmma_conv<256, 3, 1, 0, 0, 1, 48, 36, TLEN, 1, 2>
        <<<dim3(TLEN / 32, BATCH), 256, 0, stream>>>(
        P, wdecout, dec_out_b, nullptr, nullptr, nullptr, nullptr, nullptr,
        out + XHAT_OFF, TLEN, 1, 0);

    // ---- losses ----
    loss_rows<<<256, 256, 0, stream>>>(x, out + XHAT_OFF, hand_mask, lpart);
    loss_final<<<1, 256, 0, stream>>>(lpart, out + SCAL_OFF);
}
